// UNetCrossAttention_16406775071032
// MI455X (gfx1250) — matrix operand order repaired and checked by reading, not compiled, not run on hardware
//
#include <hip/hip_runtime.h>
#include <hip/hip_bf16.h>
#include <math.h>

// ---- problem constants ----
#define B_    2
#define NQ_   256
#define NK_   2048
#define DIM_  256
#define NH_   8
#define HD_   32
#define RPED_ 128

typedef __attribute__((ext_vector_type(16))) _Float16 v16h;
typedef __attribute__((ext_vector_type(8)))  _Float16 v8h;
typedef __attribute__((ext_vector_type(8)))  float    v8f;
typedef __attribute__((ext_vector_type(2)))  float    v2f;
typedef __attribute__((ext_vector_type(4)))  unsigned int v4u;
typedef __attribute__((ext_vector_type(8)))  int      v8i;
typedef __attribute__((ext_vector_type(4)))  int      v4i;

#if defined(__gfx1250__) && defined(__has_builtin)
#if __has_builtin(__builtin_amdgcn_tensor_load_to_lds) && __has_builtin(__builtin_amdgcn_s_wait_tensorcnt)
#define USE_TDM 1
#endif
#endif

// ---------------------------------------------------------------------------
// Kernel 1: build CPB tables.  tables[(i*1000 + z*100+y*10+x)*8 + h]
// MLP input at voxel (p,q,r) is (lin[p], lin[q], lin[r]), lin=linspace(-4,4,10)
// ---------------------------------------------------------------------------
__global__ void build_tables_k(const float* __restrict__ w1,  // [8,3,128]
                               const float* __restrict__ b1,  // [8,128]
                               const float* __restrict__ w2,  // [8,128,8]
                               float* __restrict__ tables) {  // [8,1000,8]
  const int i = blockIdx.x;
  const float* W1 = w1 + i * 3 * RPED_;
  const float* B1 = b1 + i * RPED_;
  const float* W2 = w2 + i * RPED_ * NH_;
  const float step = 8.0f / 9.0f;
  for (int v = threadIdx.x; v < 1000; v += blockDim.x) {
    const int p = v / 100, q = (v / 10) % 10, r = v % 10;
    const float c0 = -4.0f + p * step;   // D axis (input c=0)
    const float c1 = -4.0f + q * step;   // H axis (c=1)
    const float c2 = -4.0f + r * step;   // W axis (c=2)
    float acc[NH_];
#pragma unroll
    for (int h = 0; h < NH_; ++h) acc[h] = 0.0f;
    for (int d = 0; d < RPED_; ++d) {
      float hv = fmaf(c0, W1[d], fmaf(c1, W1[RPED_ + d], fmaf(c2, W1[2 * RPED_ + d], B1[d])));
      hv = fmaxf(hv, 0.0f);
#pragma unroll
      for (int h = 0; h < NH_; ++h) acc[h] = fmaf(hv, W2[d * NH_ + h], acc[h]);
    }
    float* o = tables + (i * 1000 + v) * NH_;
#pragma unroll
    for (int h = 0; h < NH_; ++h) o[h] = acc[h];
  }
}

// ---------------------------------------------------------------------------
// TDM: DMA one 8000-float CPB table (32 KB, 1-D tile) from global into LDS.
// D# per CDNA5 ISA ch.8: group0 = {count|flags, lds_addr, global_addr, type},
// group1 = dims/strides (data_size=4B, tensor_dim0=tile_dim0=8000).
// ---------------------------------------------------------------------------
#ifdef USE_TDM
__device__ __forceinline__ void tdm_load_table(const float* gsrc, unsigned int lds_off) {
  unsigned long long ga = (unsigned long long)(unsigned long)gsrc;
  v4u g0;
  g0[0] = 1u;                                    // count=1, user mode, no gather
  g0[1] = lds_off;                               // lds_addr (bytes)
  g0[2] = (unsigned int)(ga & 0xffffffffu);      // global_addr[31:0]
  g0[3] = (unsigned int)((ga >> 32) & 0x01ffffffu) | (2u << 30);  // addr[56:32] | type=2
  const unsigned int dim0 = 8000u;
  v8i g1;
  g1[0] = (int)(2u << 16);                       // workgroup_mask=0, data_size=2 (4B)
  g1[1] = (int)((dim0 & 0xffffu) << 16);         // tensor_dim0[15:0]  @ bits 63:48
  g1[2] = (int)(((dim0 >> 16) & 0xffffu) | (1u << 16));  // tensor_dim0[31:16], tensor_dim1=1
  g1[3] = (int)((dim0 & 0xffffu) << 16);         // tensor_dim1 hi=0, tile_dim0=8000
  g1[4] = 1;                                     // tile_dim1=1, tile_dim2=0 (unused)
  g1[5] = (int)dim0;                             // tensor_dim0_stride[31:0]
  g1[6] = 0;                                     // stride0 hi, tensor_dim1_stride lo
  g1[7] = 0;
  v4i g2; g2[0] = 1; g2[1] = 1; g2[2] = 0; g2[3] = 0;   // tensor_dim2=1, tensor_dim3=1
  v4i g3; g3[0] = 0; g3[1] = 0; g3[2] = 0; g3[3] = 0;
#if __clang_major__ >= 23
  v8i z8; for (int j = 0; j < 8; ++j) z8[j] = 0;
  __builtin_amdgcn_tensor_load_to_lds(g0, g1, g2, g3, z8, 0);
#else
  __builtin_amdgcn_tensor_load_to_lds(g0, g1, g2, g3, 0);
#endif
}
#endif

// ---------------------------------------------------------------------------
// Kernel 2: RPE via trilinear sampling, tables staged per-group into LDS
// (TDM async-tensor DMA when available). Writes attn[((b*8+h)*nQ+q)*nK + k].
// ---------------------------------------------------------------------------
__device__ __forceinline__ float grid_coord(float d) {
  float s = copysignf(1.0f, d);
  return s * __log2f(fabsf(d) * 512.0f + 1.0f) * (1.0f / 12.0f);  // /LOG2_8/MAX_VALUE
}

__global__ void rpe_k(const float* __restrict__ refp,   // [B,nQ,8,3]
                      const float* __restrict__ xyz,    // [B,nK,3]
                      const float* __restrict__ tables, // [8,1000,8]
                      float* __restrict__ attn) {       // [B,8,nQ,nK]
  __shared__ float tab[8000];
  const int tid = blockIdx.x * 256 + threadIdx.x;  // 0 .. B*nQ*nK-1
  const int k = tid & (NK_ - 1);
  const int q = (tid >> 11) & (NQ_ - 1);
  const int b = tid >> 19;
  const float kx = xyz[(b * NK_ + k) * 3 + 0];
  const float ky = xyz[(b * NK_ + k) * 3 + 1];
  const float kz = xyz[(b * NK_ + k) * 3 + 2];
  float acc[NH_];
#pragma unroll
  for (int h = 0; h < NH_; ++h) acc[h] = 0.0f;

  for (int i = 0; i < 8; ++i) {
    __syncthreads();   // previous group's reads complete before overwrite
#ifdef USE_TDM
    if ((threadIdx.x >> 5) == 0) {   // wave 0 issues the DMA (EXEC-independent)
      tdm_load_table(tables + i * 8000, (unsigned int)(unsigned long)(&tab[0]));
      __builtin_amdgcn_s_wait_tensorcnt(0);
    }
#else
    for (int t = threadIdx.x; t < 8000; t += 256) tab[t] = tables[i * 8000 + t];
#endif
    __syncthreads();

    const float* rp = refp + ((b * NQ_ + q) * 8 + i) * 3;
    const float gx = grid_coord(rp[0] - kx);  // -> W
    const float gy = grid_coord(rp[1] - ky);  // -> H
    const float gz = grid_coord(rp[2] - kz);  // -> D
    const float xf = (gx + 1.0f) * 4.5f;
    const float yf = (gy + 1.0f) * 4.5f;
    const float zf = (gz + 1.0f) * 4.5f;
    const float x0f = floorf(xf), y0f = floorf(yf), z0f = floorf(zf);
    const float wx1 = xf - x0f, wy1 = yf - y0f, wz1 = zf - z0f;
    const float wx0 = 1.0f - wx1, wy0 = 1.0f - wy1, wz0 = 1.0f - wz1;
    const int x0 = (int)x0f, y0 = (int)y0f, z0 = (int)z0f;
    const int x1 = x0 + 1, y1 = y0 + 1, z1 = z0 + 1;

    auto corner = [&](int zi, int yi, int xi, float w) {
      if ((unsigned)zi < 10u && (unsigned)yi < 10u && (unsigned)xi < 10u) {
        const float* tp = &tab[(zi * 100 + yi * 10 + xi) * NH_];
#pragma unroll
        for (int h = 0; h < NH_; ++h) acc[h] = fmaf(w, tp[h], acc[h]);
      }
    };
    corner(z0, y0, x0, wz0 * wy0 * wx0);
    corner(z0, y0, x1, wz0 * wy0 * wx1);
    corner(z0, y1, x0, wz0 * wy1 * wx0);
    corner(z0, y1, x1, wz0 * wy1 * wx1);
    corner(z1, y0, x0, wz1 * wy0 * wx0);
    corner(z1, y0, x1, wz1 * wy0 * wx1);
    corner(z1, y1, x0, wz1 * wy1 * wx0);
    corner(z1, y1, x1, wz1 * wy1 * wx1);
  }
  const long base = ((long)(b * NH_) * NQ_ + q) * NK_ + k;
#pragma unroll
  for (int h = 0; h < NH_; ++h) attn[base + (long)h * NQ_ * NK_] = acc[h];
}

// ---------------------------------------------------------------------------
// Kernel 3: f32 WMMA GEMM, compile-time N/K/gather so strides fold to
// immediates.  C = gather(in) @ w + bias; row r <-> input row (r%NR)*NB+r/NR
// when GIN (undo (seq,batch,dim) layout); same mapping on output when GOUT.
// One wave = one 16x16 tile, K stepped by 4 (v_wmma_f32_16x16x4_f32).
// ---------------------------------------------------------------------------
template <int N, int K, bool GIN, bool GOUT, bool DO16>
__global__ void gemm_wmma_f32_t(const float* __restrict__ in, const float* __restrict__ w,
                                const float* __restrict__ bias,
                                float* __restrict__ out32, _Float16* __restrict__ out16,
                                float scale16, int M, int NR, int NB) {
  const int gw = blockIdx.x * (blockDim.x >> 5) + (threadIdx.x >> 5);
  constexpr int nt = N >> 4;
  if (gw >= (M >> 4) * nt) return;           // wave-uniform: EXEC stays full
  const int n0 = (gw % nt) * 16, m0 = (gw / nt) * 16;
  const int lane = threadIdx.x & 31;
  const int ml = lane & 15;
  const int kb = (lane < 16) ? 0 : 2;        // 32-bit A/B: K split 0,1 | 2,3
  const int arow = m0 + ml;
  const int asrc = GIN ? ((arow % NR) * NB + arow / NR) : arow;
  const float* A = in + (long)asrc * K;
  const int n = n0 + ml;
  const float* W = w + n;
  v8f c = {};
#pragma unroll 4
  for (int kk = 0; kk < K; kk += 4) {
    v2f a = *(const v2f*)(A + kk + kb);
    v2f bv; bv.x = W[(kk + kb) * N]; bv.y = W[(kk + kb + 1) * N];
    c = __builtin_amdgcn_wmma_f32_16x16x4_f32(false, a, false, bv, (short)0, c, false, false);
  }
  const float bb = bias[n];
#pragma unroll
  for (int r = 0; r < 8; ++r) {
    const int m = m0 + r + ((lane < 16) ? 0 : 8);
    const int dr = GOUT ? ((m % NR) * NB + m / NR) : m;
    const float val = c[r] + bb;
    if constexpr (DO16) out16[(long)dr * N + n] = (_Float16)(val * scale16);
    else                out32[(long)dr * N + n] = val;
  }
}

// ---------------------------------------------------------------------------
// Kernel 4: attn[b,h] = qf[b,:,h*32:(h+1)*32] @ kf[b]^T + rpe(C), one
// v_wmma_f32_16x16x32_f16 per 16x16 tile (K == HEAD_DIM == 32).
// ---------------------------------------------------------------------------
__global__ void qk_wmma_k(const _Float16* __restrict__ qf,  // [B,nQ,DIM] (pre-scaled)
                          const _Float16* __restrict__ kf,  // [B,nK,HD]  (N x K storage)
                          float* __restrict__ attn) {       // [B,8,nQ,nK] (in: rpe)
  const int gw = blockIdx.x * 8 + (threadIdx.x >> 5);  // 32768 tiles
  const int kt = gw & 127;
  const int qt = (gw >> 7) & 15;
  const int h  = (gw >> 11) & 7;
  const int b  = gw >> 14;
  const int lane = threadIdx.x & 31;
  const int ml = lane & 15;

  // A: 16x32 f16. lanes<16: K 0-7 & 16-23 ; lanes>=16: K 8-15 & 24-31
  const _Float16* ap = qf + ((long)(b * NQ_ + qt * 16 + ml) * DIM_) + h * HD_ + ((lane < 16) ? 0 : 8);
  v8h alo = *(const v8h*)ap;          // 16B aligned
  v8h ahi = *(const v8h*)(ap + 16);
  v16h a = __builtin_shufflevector(alo, ahi, 0, 1, 2, 3, 4, 5, 6, 7,
                                   8, 9, 10, 11, 12, 13, 14, 15);

  // B: 32x16 f16, column n = ml; lanes<16: K 0-7 & 16-23, lanes>=16: K 8-15 & 24-31.
  // kf is stored N-major with K contiguous -> two 16-byte runs per lane (bp+0..7, bp+16..23).
  const _Float16* bp = kf + ((long)(b * NK_ + kt * 16 + ml) * HD_) + ((lane < 16) ? 0 : 8);
  v8h blo = *(const v8h*)bp;
  v8h bhi = *(const v8h*)(bp + 16);
  v16h bv = __builtin_shufflevector(blo, bhi, 0, 1, 2, 3, 4, 5, 6, 7,
                                    8, 9, 10, 11, 12, 13, 14, 15);

  // C: rpe tile from d_out attn region
  float* cp = attn + (((long)(b * NH_ + h) * NQ_ + qt * 16 + ((lane < 16) ? 0 : 8)) * NK_) + kt * 16 + ml;
  v8f c;
#pragma unroll
  for (int r = 0; r < 8; ++r) c[r] = cp[(long)r * NK_];

  c = __builtin_amdgcn_wmma_f32_16x16x32_f16(false, a, false, bv, (short)0, c, false, false);

#pragma unroll
  for (int r = 0; r < 8; ++r) cp[(long)r * NK_] = c[r];
}

// ---------------------------------------------------------------------------
// Kernel 5: row softmax over nK=2048, in place. One block per row.
// ---------------------------------------------------------------------------
__global__ void softmax_k(float* __restrict__ attn) {
  __shared__ float red[256];
  const int tid = threadIdx.x;
  float* p = attn + (long)blockIdx.x * NK_;
  float mx = -3.402823e38f;
  for (int i = tid; i < NK_; i += 256) mx = fmaxf(mx, p[i]);
  red[tid] = mx; __syncthreads();
  for (int s = 128; s > 0; s >>= 1) { if (tid < s) red[tid] = fmaxf(red[tid], red[tid + s]); __syncthreads(); }
  mx = red[0]; __syncthreads();
  float sum = 0.0f;
  for (int i = tid; i < NK_; i += 256) { float e = __expf(p[i] - mx); p[i] = e; sum += e; }
  red[tid] = sum; __syncthreads();
  for (int s = 128; s > 0; s >>= 1) { if (tid < s) red[tid] += red[tid + s]; __syncthreads(); }
  const float inv = 1.0f / red[0];
  for (int i = tid; i < NK_; i += 256) p[i] *= inv;
}

// ---------------------------------------------------------------------------
// Kernel 6: x[b,q,h*32+n] = sum_k attn[b,h,q,k] * v[b,k,n]  (f32 WMMA, K=2048)
// ---------------------------------------------------------------------------
__global__ void av_wmma_k(const float* __restrict__ attn, const float* __restrict__ v,
                          float* __restrict__ x) {  // [B,nQ,DIM]
  const int gw = blockIdx.x * 8 + (threadIdx.x >> 5);  // 512 tiles
  const int ntl = gw & 1;
  const int qt  = (gw >> 1) & 15;
  const int h   = (gw >> 5) & 7;
  const int b   = gw >> 8;
  const int lane = threadIdx.x & 31;
  const int ml = lane & 15;
  const int kb = (lane < 16) ? 0 : 2;
  const float* A = attn + (((long)(b * NH_ + h) * NQ_ + qt * 16 + ml) * NK_);
  const int n = ntl * 16 + ml;
  const float* V = v + (long)b * NK_ * HD_ + n;
  v8f c = {};
#pragma unroll 4
  for (int kk = 0; kk < NK_; kk += 4) {
    v2f a = *(const v2f*)(A + kk + kb);
    v2f bv; bv.x = V[(kk + kb) * HD_]; bv.y = V[(kk + kb + 1) * HD_];
    c = __builtin_amdgcn_wmma_f32_16x16x4_f32(false, a, false, bv, (short)0, c, false, false);
  }
#pragma unroll
  for (int r = 0; r < 8; ++r)
    x[((long)(b * NQ_ + qt * 16 + r + ((lane < 16) ? 0 : 8))) * DIM_ + h * HD_ + n] = c[r];
}

// ---------------------------------------------------------------------------
extern "C" void kernel_launch(void* const* d_in, const int* in_sizes, int n_in,
                              void* d_out, int out_size, void* d_ws, size_t ws_size,
                              hipStream_t stream) {
  const float* query  = (const float*)d_in[0];   // [nQ,B,DIM]
  const float* key    = (const float*)d_in[1];   // [nK,B,DIM]
  const float* refp   = (const float*)d_in[2];   // [B,nQ,8,3]
  /* d_in[3] reference_angle: unused by reference */
  const float* xyz    = (const float*)d_in[4];   // [B,nK,3]
  const float* q_w    = (const float*)d_in[5];
  const float* q_b    = (const float*)d_in[6];
  const float* k_w    = (const float*)d_in[7];
  const float* k_b    = (const float*)d_in[8];
  const float* v_w    = (const float*)d_in[9];
  const float* v_b    = (const float*)d_in[10];
  const float* proj_w = (const float*)d_in[11];
  const float* proj_b = (const float*)d_in[12];
  const float* cpb_w1 = (const float*)d_in[13];
  const float* cpb_b1 = (const float*)d_in[14];
  const float* cpb_w2 = (const float*)d_in[15];

  char* ws = (char*)d_ws;
  float*    tables = (float*)ws;                    // 64000 f32   (256 KB slot)
  _Float16* qf     = (_Float16*)(ws + 262144);      // B*nQ*DIM f16 (256 KB)
  _Float16* kf     = (_Float16*)(ws + 524288);      // B*nK*HD  f16 (256 KB)
  float*    vf     = (float*)(ws + 786432);         // B*nK*HD  f32 (512 KB)
  float*    xf     = (float*)(ws + 1310720);        // B*nQ*DIM f32 (512 KB)

  float* out_x    = (float*)d_out;                  // [nQ,B,DIM] = 131072
  float* out_attn = out_x + NQ_ * B_ * DIM_;        // [B,8,nQ,nK]

  // 1) CPB tables
  build_tables_k<<<8, 128, 0, stream>>>(cpb_w1, cpb_b1, cpb_w2, tables);
  // 2) RPE logits base -> out_attn (TDM-staged tables)
  rpe_k<<<(B_ * NQ_ * NK_) / 256, 256, 0, stream>>>(refp, xyz, tables, out_attn);
  // 3) projections (f32 WMMA); q pre-scaled by HEAD_DIM^-0.5 into f16
  const float qscale = 0.17677669529663687f;  // 32^-0.5
  gemm_wmma_f32_t<256, 256, true, false, true><<<64, 256, 0, stream>>>(
      query, q_w, q_b, nullptr, qf, qscale, 512, NQ_, B_);
  gemm_wmma_f32_t<32, 256, true, false, true><<<64, 256, 0, stream>>>(
      key, k_w, k_b, nullptr, kf, 1.0f, 4096, NK_, B_);
  gemm_wmma_f32_t<32, 256, true, false, false><<<64, 256, 0, stream>>>(
      key, v_w, v_b, vf, nullptr, 1.0f, 4096, NK_, B_);
  // 4) attn = q k^T + rpe (f16 WMMA, C fused)
  qk_wmma_k<<<4096, 256, 0, stream>>>(qf, kf, out_attn);
  // 5) softmax in place (this region is also output #2)
  softmax_k<<<B_ * NH_ * NQ_, 256, 0, stream>>>(out_attn);
  // 6) x = attn @ v (f32 WMMA)
  av_wmma_k<<<64, 256, 0, stream>>>(out_attn, vf, xf);
  // 7) out = x @ proj_w + proj_b, stored transposed to (nQ,B,DIM)
  gemm_wmma_f32_t<256, 256, false, true, false><<<64, 256, 0, stream>>>(
      xf, proj_w, proj_b, out_x, nullptr, 1.0f, 512, NQ_, B_);
}
